// UndistortedDepthAwareCrossAttention_48215302865353
// MI455X (gfx1250) — matrix operand order repaired and checked by reading, not compiled, not run on hardware
//
#include <hip/hip_runtime.h>

// ---------------------------------------------------------------------------
// CDNA5 / gfx1250 implementation. wave32, WMMA bf16 16x16x32, fp32 accumulate,
// async global->LDS staging for GEMM tiles.
// ---------------------------------------------------------------------------

typedef __bf16 bf16;
typedef __bf16 v16bf __attribute__((ext_vector_type(16)));
typedef float  v8f   __attribute__((ext_vector_type(8)));

// Problem constants
#define C1   256
#define NH   8
#define DH   32
#define H1   128   // query length per batch
#define H2   64    // key length per batch
#define W2   400   // batch count (sequence of columns)
#define HH   128   // feature map H
#define WW   128   // feature map W
#define NPIX (HH*WW)

// ---------------------------------------------------------------------------
// WMMA helpers (layouts per CDNA5 ISA 7.12.2)
// ---------------------------------------------------------------------------
__device__ __forceinline__ v8f wmma_bf16(v16bf a, v16bf b, v8f c) {
  return __builtin_amdgcn_wmma_f32_16x16x32_bf16(false, a, false, b, (short)0, c,
                                                 false, false);
}

union Frag16 {
  uint4 q[2];
  v16bf v;
};

// A fragment: 16x32 bf16. Caller passes pointer to this lane's row (m = lane&15)
// at the k-chunk base. k(j) = 16*(j>>3) + 8*half + (j&7). Both chunks are
// 16B-aligned -> two b128 loads.
__device__ __forceinline__ v16bf load_a_frag(const bf16* p) {
  const int half = (threadIdx.x >> 4) & 1;
  Frag16 u;
  u.q[0] = *(const uint4*)(p + half * 8);
  u.q[1] = *(const uint4*)(p + 16 + half * 8);
  return u.v;
}

// B fragment: 32x16 bf16, source stored as [N][K] row-major. Caller passes this
// lane's row (n = lane&15) at the k-chunk base. k(j) = 16*(j>>3) + 8*half + (j&7).
// Two 16B chunks -> two b128 loads.
__device__ __forceinline__ v16bf load_b_frag(const bf16* p) {
  const int half = (threadIdx.x >> 4) & 1;
  Frag16 u;
  u.q[0] = *(const uint4*)(p + half * 8);
  u.q[1] = *(const uint4*)(p + 16 + half * 8);
  return u.v;
}

// Async global -> LDS 16B copy (CDNA5 TDM-adjacent async path, ASYNCcnt).
// LDS destination operand is the 32-bit LDS offset (low bits of generic ptr).
__device__ __forceinline__ void async_copy_b128(const void* gsrc, void* ldst) {
  const unsigned loff = (unsigned)(uintptr_t)ldst;
  const unsigned long long ga = (unsigned long long)(uintptr_t)gsrc;
  asm volatile("global_load_async_to_lds_b128 %0, %1, off"
               :
               : "v"(loff), "v"(ga)
               : "memory");
}

__device__ __forceinline__ void async_wait0() {
  asm volatile("s_wait_asynccnt 0" ::: "memory");
}

// ---------------------------------------------------------------------------
// Weight fusion:  Fq = in_w[0:256] @ Wq  (so qq = a_seq @ Fq^T + bias_fused)
// ---------------------------------------------------------------------------
__global__ __launch_bounds__(256) void fuse_w_kernel(
    const float* __restrict__ in_w, const float* __restrict__ Wq,
    const float* __restrict__ Wk, const float* __restrict__ Wv,
    const float* __restrict__ out_w, bf16* __restrict__ Fq,
    bf16* __restrict__ Fk, bf16* __restrict__ Fv, bf16* __restrict__ Ow) {
  const int e = blockIdx.x * 256 + threadIdx.x;  // 65536 entries
  const int which = blockIdx.y;
  if (which == 3) { Ow[e] = (bf16)out_w[e]; return; }
  const int n = e >> 8, k = e & 255;
  const float* W = (which == 0) ? Wq : (which == 1) ? Wk : Wv;
  const float* row = in_w + (long)(which * 256 + n) * 256;
  float s = 0.f;
  for (int m = 0; m < 256; ++m) s += row[m] * W[m * 256 + k];
  bf16* F = (which == 0) ? Fq : (which == 1) ? Fk : Fv;
  F[e] = (bf16)s;
}

__global__ __launch_bounds__(256) void fuse_b_kernel(
    const float* __restrict__ in_w, const float* __restrict__ in_b,
    const float* __restrict__ bq, const float* __restrict__ bk,
    const float* __restrict__ bv, const float* __restrict__ out_b,
    float* __restrict__ bq2, float* __restrict__ bk2,
    float* __restrict__ bv2, float* __restrict__ ob2) {
  const int n = threadIdx.x;
  float sq = 0.f, sk = 0.f, sv = 0.f;
  for (int m = 0; m < 256; ++m) {
    sq += in_w[n * 256 + m] * bq[m];
    sk += in_w[(256 + n) * 256 + m] * bk[m];
    sv += in_w[(512 + n) * 256 + m] * bv[m];
  }
  bq2[n] = sq + in_b[n];
  bk2[n] = sk + in_b[256 + n];
  bv2[n] = sv + in_b[512 + n];
  ob2[n] = out_b[n];
}

// ---------------------------------------------------------------------------
// Polar coordinates + scatter index
// ---------------------------------------------------------------------------
__global__ __launch_bounds__(256) void coords_kernel(
    const float* __restrict__ rots, const float* __restrict__ fxs,
    const float* __restrict__ cxs, int cam, float* __restrict__ xb,
    float* __restrict__ yb, int* __restrict__ idxb) {
  const int i = blockIdx.x * 256 + threadIdx.x;
  if (i >= H1 * W2) return;
  const int hh = i / W2, w = i % W2;
  const float fx = fxs[cam], cx = cxs[cam];
  const float* R = rots + cam * 9;
  const float ang = atanf((2.0f * (float)w + 1.0f - cx) / fx);
  const float cr = R[3];    // (R90 @ rot)[0][0] = rots[1][0]
  const float sr = -R[0];   // (R90 @ rot)[1][0] = -rots[0][0]
  const float cA = cosf(ang), sA = sinf(ang);
  const float ca = cr * cA + sr * sA;
  const float sa = -sr * cA + cr * sA;
  const float rad = ((float)hh * (1.0f / 127.0f)) * 90.50966799187809f;
  const float x = 64.0f + rad * ca;
  const float y = 64.0f - rad * sa;
  xb[i] = x;
  yb[i] = y;
  const int xi = (int)fminf(fmaxf(rintf(x), 0.f), 127.f);
  const int yi = (int)fminf(fmaxf(rintf(y), 0.f), 127.f);
  idxb[i] = yi * WW + xi;
}

// ---------------------------------------------------------------------------
// Bilinear sample of current feature map -> a_seq (bf16) + pos_a
// grid (W2, H1), 256 threads = channels
// ---------------------------------------------------------------------------
__global__ __launch_bounds__(256) void sample_kernel(
    const float* __restrict__ a_cur, const float* __restrict__ xb,
    const float* __restrict__ yb, const float* __restrict__ pos_a,
    bf16* __restrict__ a_seq) {
  const int w = blockIdx.x, hh = blockIdx.y, c = threadIdx.x;
  float x = xb[hh * W2 + w];
  float y = yb[hh * W2 + w];
  x = fminf(fmaxf(x, 0.f), 127.f);
  y = fminf(fmaxf(y, 0.f), 127.f);
  const float x0f = floorf(x), y0f = floorf(y);
  const float x1f = fminf(x0f + 1.f, 127.f), y1f = fminf(y0f + 1.f, 127.f);
  const float wx = x - x0f, wy = y - y0f;
  const int x0 = (int)x0f, x1 = (int)x1f, y0 = (int)y0f, y1 = (int)y1f;
  const float* img = a_cur + (long)c * NPIX;
  const float v00 = img[y0 * WW + x0], v01 = img[y0 * WW + x1];
  const float v10 = img[y1 * WW + x0], v11 = img[y1 * WW + x1];
  const float v = v00 * (1.f - wx) * (1.f - wy) + v01 * wx * (1.f - wy) +
                  v10 * (1.f - wx) * wy + v11 * wx * wy;
  a_seq[((long)w * H1 + hh) * C1 + c] = (bf16)(v + pos_a[hh * C1 + c]);
}

// b_seq = transpose(b) + pos_b, bf16.  grid (W2, H2), 256 threads = channels
__global__ __launch_bounds__(256) void bseq_kernel(
    const float* __restrict__ b, const float* __restrict__ pos_b,
    bf16* __restrict__ b_seq) {
  const int w = blockIdx.x, h2 = blockIdx.y, c = threadIdx.x;
  const float v = b[(long)c * H2 * W2 + h2 * W2 + w] + pos_b[h2 * C1 + c];
  b_seq[((long)w * H2 + h2) * C1 + c] = (bf16)v;
}

// ---------------------------------------------------------------------------
// Generic WMMA GEMM: C[M][256] = A[M][256] @ B[256][256]^T + bias
// A bf16 row-major, B stored [N][K] (i.e. weight matrix) bf16, K=N=256.
// Block: 256 threads (8 waves), 64x256 C-tile, async A staging into LDS.
// ---------------------------------------------------------------------------
template <typename OutT>
__global__ __launch_bounds__(256) void gemm_wmma_256(
    const bf16* __restrict__ A, const bf16* __restrict__ Bw,
    const float* __restrict__ bias, OutT* __restrict__ C) {
  __shared__ bf16 smA[64 * 256];  // 32 KB
  const int tid = threadIdx.x;
  const long m0 = (long)blockIdx.x * 64;

  // async A-tile stage: 64x256 bf16 = 2048 x 16B, 8 per thread
  {
    const char* gA = (const char*)(A + m0 * 256);
    char* lA = (char*)smA;
#pragma unroll
    for (int i = 0; i < 8; ++i)
      async_copy_b128(gA + (size_t)(tid + i * 256) * 16,
                      lA + (size_t)(tid + i * 256) * 16);
  }
  // speculative prefetch of the likely-next A tile (global_prefetch_b8)
  __builtin_prefetch(A + (m0 + 64) * 256 + tid * 64, 0, 1);
  async_wait0();
  __syncthreads();

  const int wv = tid >> 5;
  const int lane = tid & 31;
  const int nlo = lane & 15;
  const int half = lane >> 4;
  const int n0 = wv * 32;

  v8f acc[4][2];
#pragma unroll
  for (int nt = 0; nt < 2; ++nt) {
    const float bv = bias[n0 + nt * 16 + nlo];
#pragma unroll
    for (int mt = 0; mt < 4; ++mt)
#pragma unroll
      for (int e = 0; e < 8; ++e) acc[mt][nt][e] = bv;
  }

  for (int kc = 0; kc < 8; ++kc) {
    v16bf af[4];
#pragma unroll
    for (int mt = 0; mt < 4; ++mt)
      af[mt] = load_a_frag(smA + (mt * 16 + nlo) * 256 + kc * 32);
#pragma unroll
    for (int nt = 0; nt < 2; ++nt) {
      const v16bf bfr =
          load_b_frag(Bw + (long)(n0 + nt * 16 + nlo) * 256 + kc * 32);
#pragma unroll
      for (int mt = 0; mt < 4; ++mt)
        acc[mt][nt] = wmma_bf16(af[mt], bfr, acc[mt][nt]);
    }
  }

#pragma unroll
  for (int mt = 0; mt < 4; ++mt)
#pragma unroll
    for (int nt = 0; nt < 2; ++nt) {
      const long row = m0 + mt * 16 + half * 8;
      const int col = n0 + nt * 16 + nlo;
#pragma unroll
      for (int r = 0; r < 8; ++r)
        C[(row + r) * 256 + col] = (OutT)acc[mt][nt][r];
    }
}

// ---------------------------------------------------------------------------
// Attention per (head, batch): att = qq_h @ kk_h^T / sqrt(dh); softmax;
// o_h = P @ vv_h. All intermediates in LDS. grid (NH, W2), 256 threads.
// ---------------------------------------------------------------------------
__global__ __launch_bounds__(256) void attn_kernel(
    const bf16* __restrict__ qq, const bf16* __restrict__ kk,
    const bf16* __restrict__ vv, bf16* __restrict__ o) {
  __shared__ float smAtt[H1][H2];   // 32 KB
  __shared__ bf16 smP[H1][H2];      // 16 KB
  __shared__ bf16 smVt[DH][H2];     // 4 KB  (V^T: [d][l])

  const int head = blockIdx.x;
  const long w = blockIdx.y;
  const int tid = threadIdx.x;

  const bf16* qh = qq + (w * H1) * C1 + head * DH;
  const bf16* kh = kk + (w * H2) * C1 + head * DH;
  const bf16* vh = vv + (w * H2) * C1 + head * DH;
  bf16* oh = o + (w * H1) * C1 + head * DH;

  // stage V^T into LDS: strided global gather, packed b128 LDS store
  {
    const unsigned short* vs = (const unsigned short*)vh;
    const int d = tid >> 3;         // 0..31
    const int l0 = (tid & 7) * 8;   // 0..56
    unsigned short t[8];
#pragma unroll
    for (int j = 0; j < 8; ++j) t[j] = vs[(l0 + j) * C1 + d];
    uint4 u;
    u.x = (unsigned)t[0] | ((unsigned)t[1] << 16);
    u.y = (unsigned)t[2] | ((unsigned)t[3] << 16);
    u.z = (unsigned)t[4] | ((unsigned)t[5] << 16);
    u.w = (unsigned)t[6] | ((unsigned)t[7] << 16);
    *(uint4*)&smVt[d][l0] = u;
  }

  const int wv = tid >> 5;
  const int lane = tid & 31;
  const int nlo = lane & 15;
  const int half = lane >> 4;

  // att = Q @ K^T : M=128 (wave = m-tile), N=64 (4 n-tiles), K=32 (1 chunk)
  const v16bf af = load_a_frag(qh + (long)(wv * 16 + nlo) * C1);
  v8f accs[4];
#pragma unroll
  for (int nt = 0; nt < 4; ++nt) {
    v8f acc;
#pragma unroll
    for (int e = 0; e < 8; ++e) acc[e] = 0.f;
    const v16bf bfr = load_b_frag(kh + (long)(nt * 16 + nlo) * C1);
    accs[nt] = wmma_bf16(af, bfr, acc);
  }
  const float scale = 0.17677669529663689f;  // 1/sqrt(32)
#pragma unroll
  for (int nt = 0; nt < 4; ++nt)
#pragma unroll
    for (int r = 0; r < 8; ++r)
      smAtt[wv * 16 + half * 8 + r][nt * 16 + nlo] = accs[nt][r] * scale;
  __syncthreads();

  // softmax: one thread per row
  if (tid < H1) {
    float mx = -1e30f;
    for (int j = 0; j < H2; ++j) mx = fmaxf(mx, smAtt[tid][j]);
    float s = 0.f;
    for (int j = 0; j < H2; ++j) {
      const float e = expf(smAtt[tid][j] - mx);
      smAtt[tid][j] = e;
      s += e;
    }
    const float inv = 1.f / s;
    for (int j = 0; j < H2; ++j) smP[tid][j] = (bf16)(smAtt[tid][j] * inv);
  }
  __syncthreads();

  // o = P @ V : M=128 (wave = m-tile), N=32 (2 n-tiles), K=64 (2 chunks)
  v8f occ[2];
#pragma unroll
  for (int nt = 0; nt < 2; ++nt)
#pragma unroll
    for (int e = 0; e < 8; ++e) occ[nt][e] = 0.f;
  for (int kc = 0; kc < 2; ++kc) {
    const v16bf pa = load_a_frag(&smP[wv * 16 + nlo][kc * 32]);
#pragma unroll
    for (int nt = 0; nt < 2; ++nt) {
      const v16bf vb = load_b_frag(&smVt[nt * 16 + nlo][kc * 32]);
      occ[nt] = wmma_bf16(pa, vb, occ[nt]);
    }
  }
#pragma unroll
  for (int nt = 0; nt < 2; ++nt)
#pragma unroll
    for (int r = 0; r < 8; ++r)
      oh[(long)(wv * 16 + half * 8 + r) * C1 + nt * 16 + nlo] =
          (bf16)occ[nt][r];
}

// ---------------------------------------------------------------------------
// Scatter / normalize / apply
// ---------------------------------------------------------------------------
__global__ __launch_bounds__(256) void zero_kernel(float* __restrict__ rest,
                                                   float* __restrict__ cnt) {
  const long i = (long)blockIdx.x * 256 + threadIdx.x;
  if (i < (long)C1 * NPIX)
    rest[i] = 0.f;
  else if (i < (long)C1 * NPIX + NPIX)
    cnt[i - (long)C1 * NPIX] = 0.f;
}

__global__ __launch_bounds__(256) void scatter_kernel(
    const float* __restrict__ a_enh, const int* __restrict__ idxb,
    float* __restrict__ rest, float* __restrict__ cnt) {
  const int w = blockIdx.x, hh = blockIdx.y, c = threadIdx.x;
  const int id = idxb[hh * W2 + w];
  const float v = a_enh[((long)w * H1 + hh) * C1 + c];
  atomicAdd(&rest[(long)c * NPIX + id], v);
  if (c == 0) atomicAdd(&cnt[id], 1.0f);
}

__global__ __launch_bounds__(256) void apply_kernel(
    const float* __restrict__ rest, const float* __restrict__ cnt,
    float* __restrict__ a_cur, float* __restrict__ out, int last) {
  const long p = (long)blockIdx.x * 256 + threadIdx.x;  // C1*NPIX total
  float cn = cnt[p & (NPIX - 1)];
  cn = (cn == 0.f) ? 1.f : cn;
  const float e = rest[p] / cn;
  const float an = a_cur[p] + e;
  a_cur[p] = an;
  if (last) out[p] = an + e;  // a_final + enh_last
}

// ---------------------------------------------------------------------------
// Host orchestration
// ---------------------------------------------------------------------------
extern "C" void kernel_launch(void* const* d_in, const int* in_sizes, int n_in,
                              void* d_out, int out_size, void* d_ws,
                              size_t ws_size, hipStream_t stream) {
  const float* list_a = (const float*)d_in[0];
  const float* list_b = (const float*)d_in[1];
  const float* rots = (const float*)d_in[3];
  const float* fxs = (const float*)d_in[5];
  const float* cxs = (const float*)d_in[6];
  const float* pos_a = (const float*)d_in[7];
  const float* pos_b = (const float*)d_in[8];
  const float* Wq = (const float*)d_in[9];
  const float* bq = (const float*)d_in[10];
  const float* Wk = (const float*)d_in[11];
  const float* bk = (const float*)d_in[12];
  const float* Wv = (const float*)d_in[13];
  const float* bv = (const float*)d_in[14];
  const float* in_w = (const float*)d_in[15];
  const float* in_b = (const float*)d_in[16];
  const float* out_w = (const float*)d_in[17];
  const float* out_b = (const float*)d_in[18];
  float* out = (float*)d_out;

  // workspace layout (all sizes 256B-aligned)
  char* ws = (char*)d_ws;
  size_t off = 0;
  auto alloc = [&](size_t bytes) {
    void* p = ws + off;
    off += (bytes + 255) & ~(size_t)255;
    return p;
  };
  float* a_cur = (float*)alloc((size_t)C1 * NPIX * 4);          // 16 MB
  float* xbuf = (float*)alloc((size_t)H1 * W2 * 4);
  float* ybuf = (float*)alloc((size_t)H1 * W2 * 4);
  int* idxb = (int*)alloc((size_t)H1 * W2 * 4);
  bf16* Fq = (bf16*)alloc((size_t)C1 * C1 * 2);
  bf16* Fk = (bf16*)alloc((size_t)C1 * C1 * 2);
  bf16* Fv = (bf16*)alloc((size_t)C1 * C1 * 2);
  bf16* Ow = (bf16*)alloc((size_t)C1 * C1 * 2);
  float* bq2 = (float*)alloc(C1 * 4);
  float* bk2 = (float*)alloc(C1 * 4);
  float* bv2 = (float*)alloc(C1 * 4);
  float* ob2 = (float*)alloc(C1 * 4);
  bf16* a_seq = (bf16*)alloc((size_t)W2 * H1 * C1 * 2);  // reused for o
  bf16* qq = (bf16*)alloc((size_t)W2 * H1 * C1 * 2);
  bf16* kkb = (bf16*)alloc((size_t)W2 * H2 * C1 * 2);
  bf16* vvb = (bf16*)alloc((size_t)W2 * H2 * C1 * 2);
  float* a_enh = (float*)alloc((size_t)W2 * H1 * C1 * 4);
  float* rest = (float*)alloc((size_t)C1 * NPIX * 4);
  float* cnt = (float*)alloc((size_t)NPIX * 4);
  (void)ws_size; (void)n_in; (void)in_sizes; (void)out_size;

  // a_cur = list_a[0]
  hipMemcpyAsync(a_cur, list_a, (size_t)C1 * NPIX * 4,
                 hipMemcpyDeviceToDevice, stream);

  // fuse projection weights (once; shared by both cameras)
  fuse_w_kernel<<<dim3(256, 4), 256, 0, stream>>>(in_w, Wq, Wk, Wv, out_w, Fq,
                                                  Fk, Fv, Ow);
  fuse_b_kernel<<<1, 256, 0, stream>>>(in_w, in_b, bq, bk, bv, out_b, bq2, bk2,
                                       bv2, ob2);

  const int cams[2] = {0, 3};
  for (int ci = 0; ci < 2; ++ci) {
    const int cam = cams[ci];
    const int last = (ci == 1);

    coords_kernel<<<(H1 * W2 + 255) / 256, 256, 0, stream>>>(
        rots, fxs, cxs, cam, xbuf, ybuf, idxb);
    sample_kernel<<<dim3(W2, H1), 256, 0, stream>>>(a_cur, xbuf, ybuf, pos_a,
                                                    a_seq);

    // b_seq staged in the (currently free) a_enh region; a_enh is fully
    // overwritten later by the out-projection GEMM.
    bf16* b_seq = (bf16*)a_enh;
    bseq_kernel<<<dim3(W2, H2), 256, 0, stream>>>(
        list_b + (size_t)cam * C1 * H2 * W2, pos_b, b_seq);

    // fused projections: qq = a_seq @ Fq^T + bq2 ; kk/vv from b_seq
    gemm_wmma_256<bf16><<<(W2 * H1) / 64, 256, 0, stream>>>(a_seq, Fq, bq2, qq);
    gemm_wmma_256<bf16><<<(W2 * H2) / 64, 256, 0, stream>>>(b_seq, Fk, bk2, kkb);
    gemm_wmma_256<bf16><<<(W2 * H2) / 64, 256, 0, stream>>>(b_seq, Fv, bv2, vvb);

    // attention; o reuses a_seq storage (a_seq no longer needed)
    bf16* o = a_seq;
    attn_kernel<<<dim3(NH, W2), 256, 0, stream>>>(qq, kkb, vvb, o);

    // output projection -> a_enh (fp32); overwrites the b_seq staging alias
    gemm_wmma_256<float><<<(W2 * H1) / 64, 256, 0, stream>>>(o, Ow, ob2, a_enh);

    // scatter-add with count normalization; update a_cur (and out on last)
    zero_kernel<<<((C1 * NPIX + NPIX) + 255) / 256, 256, 0, stream>>>(rest,
                                                                      cnt);
    scatter_kernel<<<dim3(W2, H1), 256, 0, stream>>>(a_enh, idxb, rest, cnt);
    apply_kernel<<<(C1 * NPIX) / 256, 256, 0, stream>>>(rest, cnt, a_cur, out,
                                                        last);
  }
}
